// LiquidNCPNetwork_82995948028243
// MI455X (gfx1250) — matrix operand order repaired and checked by reading, not compiled, not run on hardware
//
#include <hip/hip_runtime.h>
#include <math.h>

// ---------------- CDNA5 wave32 WMMA types ----------------
typedef __bf16 v16bf __attribute__((ext_vector_type(16)));
typedef __bf16 v8bf  __attribute__((ext_vector_type(8)));
typedef float  v8f   __attribute__((ext_vector_type(8)));

// ---------------- problem constants ----------------
static constexpr int SEQB    = 512;   // SEQ
static constexpr int BATCH   = 512;
static constexpr int OUTF    = 64;
static constexpr int NUNITS  = 320;   // 154 + 102 + 64
static constexpr int THREADS = 320;   // 10 wave32s: layer0's 10 N-tiles in ONE round
// layer 0: in 256, hid 154, head 410 -> Kpad 416 (13 KT), Npad 160 (10 NT)
// layer 1: in 154, hid 102, head 256 -> Kpad 256 ( 8 KT), Npad 112 ( 7 NT)
// layer 2: in 102, hid  64, head 166 -> Kpad 192 ( 6 KT), Npad  64 ( 4 NT)

// workspace layout (bytes); packed fragments are [head(5)][NT][KT][lane(32)][16 halves]
static constexpr size_t FRAG0_OFF = 0;        // 5*10*13*1024 = 665600
static constexpr size_t FRAG1_OFF = 665600;   // 5*7*8*1024  = 286720
static constexpr size_t FRAG2_OFF = 952320;   // 5*4*6*1024  = 122880
static constexpr size_t BIAS0_OFF = 1075200;  // 5*160*4
static constexpr size_t BIAS1_OFF = 1078400;  // 5*112*4
static constexpr size_t BIAS2_OFF = 1080640;  // 5*64*4   (total ~1.03 MB of d_ws)

// ================= weight prep: fold mask, cast bf16, pack into B-fragment order =================
// B fragment as packed here (A uses the same k order): lane L holds column N=L%16; lanes 0-15 K=0..15,
// lanes 16-31 K=16..31; 16 consecutive halves per lane (VGPR v <-> K=2v,2v+1).
__global__ void prep_frags(const float* __restrict__ wg, const float* __restrict__ wh,
                           const float* __restrict__ wfg, const float* __restrict__ wfh,
                           const float* __restrict__ wp, const float* __restrict__ mask,
                           unsigned short* __restrict__ out_,
                           int nhid, int head, int NT, int KT)
{
    int gid = blockIdx.x * blockDim.x + threadIdx.x;
    int total = 5 * NT * KT * 32;
    if (gid >= total) return;
    int lane = gid & 31;
    int frag = gid >> 5;
    int kt   = frag % KT;
    int tmp  = frag / KT;
    int nt   = tmp % NT;
    int hsel = tmp / NT;
    const float* W = (hsel == 0) ? wg : (hsel == 1) ? wh : (hsel == 2) ? wfg
                   : (hsel == 3) ? wfh : wp;
    int n  = nt * 16 + (lane & 15);
    int kb = kt * 32 + (lane >> 4) * 16;
    __bf16* dst = (__bf16*)out_ + ((size_t)frag * 32 + lane) * 16;
#pragma unroll
    for (int j = 0; j < 16; ++j) {
        int k = kb + j;
        float v = 0.0f;
        if (n < nhid && k < head) {
            size_t idx = (size_t)n * head + k;
            v = W[idx] * mask[idx];
        }
        dst[j] = (__bf16)v;
    }
}

__global__ void prep_bias(const float* bg, const float* bh, const float* bfg,
                          const float* bfh, const float* bp,
                          float* __restrict__ out, int nhid, int npad)
{
    int gid = blockIdx.x * blockDim.x + threadIdx.x;
    if (gid >= 5 * npad) return;
    int h = gid / npad, n = gid % npad;
    const float* B = (h == 0) ? bg : (h == 1) ? bh : (h == 2) ? bfg : (h == 3) ? bfh : bp;
    out[gid] = (n < nhid) ? B[n] : 0.0f;
}

// ================= one liquid cell layer for a 16-row batch tile =================
// One wave per N-tile (nt = wave), so every layer finishes in a single nt-round.
// 5 head GEMMs via v_wmma_f32_16x16x32_bf16 sharing one A fragment per K-tile;
// gating applied in-register; new_h staged to LDS (copied back by caller after
// barrier); y goes to the next layer's LDS x-region, or to global for layer 2.
__device__ __forceinline__ void run_layer(
    const __bf16* __restrict__ zin, int zs,
    __bf16* __restrict__ stage,
    __bf16* __restrict__ zoutx, int zos,
    float*  __restrict__ gout,                    // layer2 only (else nullptr)
    const v16bf* __restrict__ wf,
    const float* __restrict__ bias, int npad,
    int KT, int NT, int nhid, float ts,
    int lane, int wave)
{
    const int m0 = lane & 15;   // A-row / D-column index
    const int hi = lane >> 4;
    const int nt = wave;
    if (nt >= NT) return;       // wave-uniform: EXEC all-ones inside

    const v16bf* w0 = wf + (size_t)((0 * NT + nt) * KT) * 32 + lane;
    const v16bf* w1 = wf + (size_t)((1 * NT + nt) * KT) * 32 + lane;
    const v16bf* w2 = wf + (size_t)((2 * NT + nt) * KT) * 32 + lane;
    const v16bf* w3 = wf + (size_t)((3 * NT + nt) * KT) * 32 + lane;
    const v16bf* w4 = wf + (size_t)((4 * NT + nt) * KT) * 32 + lane;

    v8f ag  = {0.f,0.f,0.f,0.f,0.f,0.f,0.f,0.f};
    v8f ah  = {0.f,0.f,0.f,0.f,0.f,0.f,0.f,0.f};
    v8f afg = {0.f,0.f,0.f,0.f,0.f,0.f,0.f,0.f};
    v8f afh = {0.f,0.f,0.f,0.f,0.f,0.f,0.f,0.f};
    v8f ap  = {0.f,0.f,0.f,0.f,0.f,0.f,0.f,0.f};

    for (int kt = 0; kt < KT; ++kt) {
        // A fragment: lane holds row M=lane%16; halves K = k0..k0+15 (k0 = 16*hi)
        const __bf16* zr = zin + (size_t)m0 * zs + kt * 32 + hi * 16;
        v8bf alo  = *(const v8bf*)zr;
        v8bf ahi8 = *(const v8bf*)(zr + 8);
        v16bf a = __builtin_shufflevector(alo, ahi8,
                    0,1,2,3,4,5,6,7,8,9,10,11,12,13,14,15);
        v16bf bg  = w0[(size_t)kt * 32];
        v16bf bh  = w1[(size_t)kt * 32];
        v16bf bfg = w2[(size_t)kt * 32];
        v16bf bfh = w3[(size_t)kt * 32];
        v16bf bp  = w4[(size_t)kt * 32];
        ag  = __builtin_amdgcn_wmma_f32_16x16x32_bf16(false, a, false, bg,  (short)0, ag,  false, false);
        ah  = __builtin_amdgcn_wmma_f32_16x16x32_bf16(false, a, false, bh,  (short)0, ah,  false, false);
        afg = __builtin_amdgcn_wmma_f32_16x16x32_bf16(false, a, false, bfg, (short)0, afg, false, false);
        afh = __builtin_amdgcn_wmma_f32_16x16x32_bf16(false, a, false, bfh, (short)0, afh, false, false);
        ap  = __builtin_amdgcn_wmma_f32_16x16x32_bf16(false, a, false, bp,  (short)0, ap,  false, false);
    }

    int nglob = nt * 16 + m0;
    float bg_s  = bias[0 * npad + nglob];
    float bh_s  = bias[1 * npad + nglob];
    float bfg_s = bias[2 * npad + nglob];
    float bfh_s = bias[3 * npad + nglob];
    float bp_s  = bias[4 * npad + nglob];

#pragma unroll
    for (int i = 0; i < 8; ++i) {
        int   m    = hi * 8 + i;                 // D layout: M = vgpr + 8*hi
        float g    = tanhf(ag[i] + bg_s);
        float hh   = tanhf(ah[i] + bh_s);
        float s    = (afg[i] + bfg_s) * ts + (afh[i] + bfh_s);
        float gate = 1.0f / (1.0f + __expf(-s));
        float nh   = g * (1.0f - gate) + hh * gate;
        float y    = ap[i] + bp_s + nh;
        stage[m * 160 + nglob] = (__bf16)nh;
        if (gout) {
            gout[(size_t)m * (SEQB * OUTF) + nglob] = y;     // nglob < 64 always
        } else if (nglob < nhid) {
            zoutx[m * zos + nglob] = (__bf16)y;
        }
    }
}

// ================= main persistent-tile scan kernel =================
__global__ __launch_bounds__(THREADS)
void liquid_ncp_main(const float* __restrict__ x,
                     const float* __restrict__ h0,
                     const float* __restrict__ tspans,
                     const unsigned short* __restrict__ frag0_,
                     const unsigned short* __restrict__ frag1_,
                     const unsigned short* __restrict__ frag2_,
                     const float* __restrict__ bias0,
                     const float* __restrict__ bias1,
                     const float* __restrict__ bias2,
                     float* __restrict__ yout,
                     float* __restrict__ hfin)
{
    __shared__ __attribute__((aligned(32))) __bf16 z0[16 * 416]; // [x(256) | h0(154) | pad(6)]
    __shared__ __attribute__((aligned(32))) __bf16 z1[16 * 256]; // [y0(154) | h1(102)]
    __shared__ __attribute__((aligned(32))) __bf16 z2[16 * 192]; // [y1(102) | h2(64) | pad(26)]
    __shared__ __attribute__((aligned(32))) __bf16 stage[16 * 160];

    const int tid  = threadIdx.x;
    const int lane = tid & 31;
    const int wave = tid >> 5;
    const int b0   = blockIdx.x * 16;

    const v16bf* frag0 = (const v16bf*)frag0_;
    const v16bf* frag1 = (const v16bf*)frag1_;
    const v16bf* frag2 = (const v16bf*)frag2_;

    // init pads + load h state (zeros per reference, but do it properly)
    for (int i = tid; i < 16 * 6;  i += THREADS) z0[(i / 6)  * 416 + 410 + (i % 6)]  = (__bf16)0.0f;
    for (int i = tid; i < 16 * 26; i += THREADS) z2[(i / 26) * 192 + 166 + (i % 26)] = (__bf16)0.0f;
    for (int i = tid; i < 16 * NUNITS; i += THREADS) {
        int r = i / NUNITS, j = i % NUNITS;
        float v = h0[(size_t)(b0 + r) * NUNITS + j];
        if (j < 154)      z0[r * 416 + 256 + j] = (__bf16)v;
        else if (j < 256) z1[r * 256 + j]       = (__bf16)v;
        else              z2[r * 192 + 102 + (j - 256)] = (__bf16)v;
    }

    for (int t = 0; t < SEQB; ++t) {
        float ts = tspans[t];
        // stream this step's x tile (16 rows x 256 f32) into z0 as bf16
        const float* xb = x + ((size_t)b0 * SEQB + t) * 256;
        for (int v = tid; v < 16 * 64; v += THREADS) {
            int r = v >> 6, c = (v & 63) * 4;
            const float4 f = *(const float4*)(xb + (size_t)r * SEQB * 256 + c);
            __bf16* d = &z0[r * 416 + c];
            d[0] = (__bf16)f.x; d[1] = (__bf16)f.y; d[2] = (__bf16)f.z; d[3] = (__bf16)f.w;
        }
        if (t + 1 < SEQB && tid < 256) { // prefetch next timestep's x tile (global_prefetch_b8)
            int r = tid & 15, c = (tid >> 4) * 16;
            __builtin_prefetch(xb + 256 + (size_t)r * SEQB * 256 + c, 0, 0);
        }
        __syncthreads();

        // ---- layer 0 : 10 N-tiles on 10 waves, single round ----
        run_layer(z0, 416, stage, z1, 256, nullptr, frag0, bias0, 160, 13, 10, 154,
                  ts, lane, wave);
        __syncthreads();
        for (int i = tid; i < 16 * 154; i += THREADS) {
            int r = i / 154, c = i % 154;
            z0[r * 416 + 256 + c] = stage[r * 160 + c];
        }
        __syncthreads();

        // ---- layer 1 : 7 N-tiles ----
        run_layer(z1, 256, stage, z2, 192, nullptr, frag1, bias1, 112, 8, 7, 102,
                  ts, lane, wave);
        __syncthreads();
        for (int i = tid; i < 16 * 102; i += THREADS) {
            int r = i / 102, c = i % 102;
            z1[r * 256 + 154 + c] = stage[r * 160 + c];
        }
        __syncthreads();

        // ---- layer 2 : 4 N-tiles (y -> global) ----
        float* yo = yout + ((size_t)b0 * SEQB + t) * OUTF;
        run_layer(z2, 192, stage, nullptr, 0, yo, frag2, bias2, 64, 6, 4, 64,
                  ts, lane, wave);
        __syncthreads();
        for (int i = tid; i < 16 * 64; i += THREADS) {
            int r = i / 64, c = i % 64;
            z2[r * 192 + 102 + c] = stage[r * 160 + c];
        }
        __syncthreads();
    }

    // final hidden state
    for (int i = tid; i < 16 * NUNITS; i += THREADS) {
        int r = i / NUNITS, j = i % NUNITS;
        float v;
        if (j < 154)      v = (float)z0[r * 416 + 256 + j];
        else if (j < 256) v = (float)z1[r * 256 + j];
        else              v = (float)z2[r * 192 + 102 + (j - 256)];
        hfin[(size_t)(b0 + r) * NUNITS + j] = v;
    }
}

// ================= launch =================
extern "C" void kernel_launch(void* const* d_in, const int* in_sizes, int n_in,
                              void* d_out, int out_size, void* d_ws, size_t ws_size,
                              hipStream_t stream)
{
    (void)in_sizes; (void)n_in; (void)out_size; (void)ws_size;
    const float* x  = (const float*)d_in[0];
    const float* h0 = (const float*)d_in[1];
    const float* ts = (const float*)d_in[2];
    const float* mask[3] = {(const float*)d_in[3], (const float*)d_in[4], (const float*)d_in[5]};
    const float* W[3][5]; const float* B[3][5];
    for (int l = 0; l < 3; ++l)
        for (int h = 0; h < 5; ++h) {
            W[l][h] = (const float*)d_in[6 + l * 10 + h * 2];
            B[l][h] = (const float*)d_in[6 + l * 10 + h * 2 + 1];
        }

    char* ws = (char*)d_ws;
    unsigned short* frag0 = (unsigned short*)(ws + FRAG0_OFF);
    unsigned short* frag1 = (unsigned short*)(ws + FRAG1_OFF);
    unsigned short* frag2 = (unsigned short*)(ws + FRAG2_OFF);
    float* bias0 = (float*)(ws + BIAS0_OFF);
    float* bias1 = (float*)(ws + BIAS1_OFF);
    float* bias2 = (float*)(ws + BIAS2_OFF);

    // fold mask into weights, cast to bf16, pack B fragments (runs every call; cheap)
    prep_frags<<<(5 * 10 * 13 * 32 + 255) / 256, 256, 0, stream>>>(
        W[0][0], W[0][1], W[0][2], W[0][3], W[0][4], mask[0], frag0, 154, 410, 10, 13);
    prep_frags<<<(5 * 7 * 8 * 32 + 255) / 256, 256, 0, stream>>>(
        W[1][0], W[1][1], W[1][2], W[1][3], W[1][4], mask[1], frag1, 102, 256, 7, 8);
    prep_frags<<<(5 * 4 * 6 * 32 + 255) / 256, 256, 0, stream>>>(
        W[2][0], W[2][1], W[2][2], W[2][3], W[2][4], mask[2], frag2, 64, 166, 4, 6);
    prep_bias<<<(5 * 160 + 255) / 256, 256, 0, stream>>>(
        B[0][0], B[0][1], B[0][2], B[0][3], B[0][4], bias0, 154, 160);
    prep_bias<<<(5 * 112 + 255) / 256, 256, 0, stream>>>(
        B[1][0], B[1][1], B[1][2], B[1][3], B[1][4], bias1, 102, 112);
    prep_bias<<<(5 * 64 + 255) / 256, 256, 0, stream>>>(
        B[2][0], B[2][1], B[2][2], B[2][3], B[2][4], bias2, 64, 64);

    float* yout = (float*)d_out;
    float* hfin = yout + (size_t)BATCH * SEQB * OUTF;
    liquid_ncp_main<<<BATCH / 16, THREADS, 0, stream>>>(
        x, h0, ts, frag0, frag1, frag2, bias0, bias1, bias2, yout, hfin);
}
